// GroupQueryAttention_163208757512
// MI455X (gfx1250) — matrix operand order repaired and checked by reading, not compiled, not run on hardware
//
#include <hip/hip_runtime.h>

// ---------------------------------------------------------------------------
// MHA for B=2, S=2048, D=1024, H=16, DK=64 on gfx1250 (CDNA5, wave32, WMMA)
// Pipeline: fp32->f16 cast | QKV projections (WMMA) | flash-attention (WMMA +
// online softmax, scores computed transposed to avoid repacks) | O projection.
// ---------------------------------------------------------------------------

typedef __attribute__((ext_vector_type(16))) _Float16 v16h;
typedef __attribute__((ext_vector_type(8)))  _Float16 v8h;
typedef __attribute__((ext_vector_type(4)))  _Float16 v4h;
typedef __attribute__((ext_vector_type(8)))  float    v8f;
typedef __attribute__((ext_vector_type(4)))  float    v4f;

constexpr int B_  = 2;
constexpr int S_  = 2048;
constexpr int D_  = 1024;
constexpr int H_  = 16;
constexpr int DK_ = 64;
constexpr int M_  = B_ * S_;   // 4096 rows in projection GEMMs

// ---------------------------------------------------------------------------
// fp32 -> f16 cast, 4 elements per thread
// ---------------------------------------------------------------------------
__global__ void __launch_bounds__(256)
mha_cvt_f16(const float* __restrict__ src, _Float16* __restrict__ dst, int n4) {
  int i = blockIdx.x * blockDim.x + threadIdx.x;
  if (i < n4) {
    v4f v = ((const v4f*)src)[i];
    v4h h;
    h[0] = (_Float16)v[0]; h[1] = (_Float16)v[1];
    h[2] = (_Float16)v[2]; h[3] = (_Float16)v[3];
    ((v4h*)dst)[i] = h;
  }
}

// ---------------------------------------------------------------------------
// TN GEMM: out = A[M,K] * W[N,K]^T + bias, one 64x32 tile per wave
// (MB=4 row blocks x NB=2 col blocks of 16x16; 8 accum tiles = 64 VGPRs).
// mode 0: f16 out, scattered to [B,H,S,DK]  (Q,K projections)
// mode 1: f16 out, packed to   [B,H,DK,S]  (V projection, pre-transposed)
// mode 2: f32 out, row-major   [M,N]       (final O projection -> d_out)
// ---------------------------------------------------------------------------
__global__ void __launch_bounds__(256)
mha_gemm_tn(const _Float16* __restrict__ A,
            const _Float16* __restrict__ W,
            const float* __restrict__ bias,
            void* __restrict__ out,
            int mode, int Mdim, int Ndim, int Kdim) {
  constexpr int MB = 4, NB = 2;
  const int lane = threadIdx.x & 31;
  const int l16  = lane & 15;
  const int lhi  = lane >> 4;
  const int wave = blockIdx.x * (blockDim.x >> 5) + (threadIdx.x >> 5);
  const int mTiles = Mdim >> 6;                 // 64-row tiles
  const int mTile  = wave % mTiles;
  const int nTile  = wave / mTiles;             // 32-col tiles
  if (nTile >= (Ndim >> 5)) return;

  v8f acc[MB][NB];
#pragma unroll
  for (int i = 0; i < MB; ++i)
#pragma unroll
    for (int j = 0; j < NB; ++j)
#pragma unroll
      for (int g = 0; g < 8; ++g) acc[i][j][g] = 0.0f;

  const _Float16* aBase = A + (size_t)(mTile * 64 + l16) * Kdim;
  const _Float16* wBase = W + (size_t)(nTile * 32 + l16) * Kdim;

#pragma unroll 1
  for (int k = 0; k < Kdim; k += 32) {
    v16h aop[MB], bop[NB];
#pragma unroll
    for (int mb = 0; mb < MB; ++mb) {
      // A operand: lane-half interleaved K chunks {0..7,16..23}/{8..15,24..31}
      const _Float16* p = aBase + (size_t)mb * 16 * Kdim + k + lhi * 8;
      v8h lo = *(const v8h*)p;
      v8h hi = *(const v8h*)(p + 16);
#pragma unroll
      for (int t = 0; t < 8; ++t) { aop[mb][t] = lo[t]; aop[mb][t + 8] = hi[t]; }
    }
#pragma unroll
    for (int nb = 0; nb < NB; ++nb) {
      // B operand: lane-half interleaved K chunks {0..7,16..23}/{8..15,24..31}
      const _Float16* p = wBase + (size_t)nb * 16 * Kdim + k + lhi * 8;
      v8h lo = *(const v8h*)p;
      v8h hi = *(const v8h*)(p + 16);
#pragma unroll
      for (int t = 0; t < 8; ++t) { bop[nb][t] = lo[t]; bop[nb][t + 8] = hi[t]; }
    }
#pragma unroll
    for (int mb = 0; mb < MB; ++mb)
#pragma unroll
      for (int nb = 0; nb < NB; ++nb)
        acc[mb][nb] = __builtin_amdgcn_wmma_f32_16x16x32_f16(
            false, aop[mb], false, bop[nb], (short)0, acc[mb][nb], false, false);
  }

  // Epilogue. C layout: VGPR g, lane -> (Mrow = g + 8*lhi, Ncol = l16).
#pragma unroll
  for (int nb = 0; nb < NB; ++nb) {
    const int n  = nTile * 32 + nb * 16 + l16;
    const float bn = bias[n];
#pragma unroll
    for (int mb = 0; mb < MB; ++mb) {
      const int m0 = mTile * 64 + mb * 16 + lhi * 8;  // row of g==0
      if (mode == 2) {
        float* o32 = (float*)out;
#pragma unroll
        for (int g = 0; g < 8; ++g)
          o32[(size_t)(m0 + g) * Ndim + n] = acc[mb][nb][g] + bn;
      } else if (mode == 0) {
        _Float16* o16 = (_Float16*)out;
        const int h = n >> 6, dk = n & 63;
#pragma unroll
        for (int g = 0; g < 8; ++g) {
          const int m = m0 + g;
          const int b = m >> 11, s = m & (S_ - 1);
          o16[(((size_t)b * H_ + h) * S_ + s) * DK_ + dk] =
              (_Float16)(acc[mb][nb][g] + bn);
        }
      } else {  // mode 1: transposed V, 8 consecutive s -> one b128 store
        _Float16* o16 = (_Float16*)out;
        const int h = n >> 6, dk = n & 63;
        const int b = m0 >> 11, s0g = m0 & (S_ - 1);
        v8h pk;
#pragma unroll
        for (int g = 0; g < 8; ++g) pk[g] = (_Float16)(acc[mb][nb][g] + bn);
        *(v8h*)(o16 + (((size_t)b * H_ + h) * DK_ + dk) * S_ + s0g) = pk;
      }
    }
  }
}

// ---------------------------------------------------------------------------
// Flash attention, one wave per 16 q-rows of one (b,h).
// Computes S^T = K * Q^T so the P^T operand for O^T = V^T * P^T is formed
// from the score accumulators with a single shfl_xor(16) exchange.
// ---------------------------------------------------------------------------
__global__ void __launch_bounds__(256)
mha_attn(const _Float16* __restrict__ q,   // [B,H,S,DK]
         const _Float16* __restrict__ kk_, // [B,H,S,DK]
         const _Float16* __restrict__ vt,  // [B,H,DK,S]
         _Float16* __restrict__ xo) {      // [B,S,D]
  const int lane = threadIdx.x & 31;
  const int l16  = lane & 15;
  const int lhi  = lane >> 4;
  const int wave = blockIdx.x * (blockDim.x >> 5) + (threadIdx.x >> 5);
  const int qtPerBH = S_ / 16;                 // 128
  const int bh = wave / qtPerBH;
  const int s0 = (wave % qtPerBH) * 16;
  if (bh >= B_ * H_) return;
  const int b = bh / H_, h = bh % H_;

  const _Float16* qB = q   + (size_t)bh * S_ * DK_;
  const _Float16* kB = kk_ + (size_t)bh * S_ * DK_;
  const _Float16* vB = vt  + (size_t)bh * DK_ * S_;

  // Q^T B-operands for both 32-wide d chunks (reused for all k tiles).
  v16h qop[2];
#pragma unroll
  for (int kc = 0; kc < 2; ++kc) {
    const _Float16* p = qB + (size_t)(s0 + l16) * DK_ + kc * 32 + lhi * 8;
    v8h lo = *(const v8h*)p;
    v8h hi = *(const v8h*)(p + 16);
#pragma unroll
    for (int t = 0; t < 8; ++t) { qop[kc][t] = lo[t]; qop[kc][t + 8] = hi[t]; }
  }

  float mrun = -1e30f, lrun = 0.0f;
  v8f o[4];
#pragma unroll
  for (int t = 0; t < 4; ++t)
#pragma unroll
    for (int g = 0; g < 8; ++g) o[t][g] = 0.0f;

  const float scale = 0.125f;  // 1/sqrt(DK)

#pragma unroll 1
  for (int j0 = 0; j0 < S_; j0 += 32) {
    if (j0 + 32 < S_) {  // prefetch next K/V tiles (global_prefetch_b8)
      __builtin_prefetch(kB + (size_t)(j0 + 32 + l16) * DK_, 0, 3);
      __builtin_prefetch(vB + (size_t)l16 * S_ + j0 + 32, 0, 3);
    }
    // --- S^T tiles: st[jb] covers j in [j0+16*jb, j0+16*jb+16) ---
    v8f st[2];
#pragma unroll
    for (int jb = 0; jb < 2; ++jb)
#pragma unroll
      for (int g = 0; g < 8; ++g) st[jb][g] = 0.0f;
#pragma unroll
    for (int jb = 0; jb < 2; ++jb) {
#pragma unroll
      for (int kc = 0; kc < 2; ++kc) {
        const _Float16* p =
            kB + (size_t)(j0 + jb * 16 + l16) * DK_ + kc * 32 + lhi * 8;
        v8h lo = *(const v8h*)p;
        v8h hi = *(const v8h*)(p + 16);
        v16h a;
#pragma unroll
        for (int t = 0; t < 8; ++t) { a[t] = lo[t]; a[t + 8] = hi[t]; }
        st[jb] = __builtin_amdgcn_wmma_f32_16x16x32_f16(
            false, a, false, qop[kc], (short)0, st[jb], false, false);
      }
    }
    // --- online softmax over j (per-lane 16 values + partner lane) ---
    float tmax = -1e30f;
#pragma unroll
    for (int jb = 0; jb < 2; ++jb)
#pragma unroll
      for (int g = 0; g < 8; ++g) {
        st[jb][g] *= scale;
        tmax = fmaxf(tmax, st[jb][g]);
      }
    tmax = fmaxf(tmax, __shfl_xor(tmax, 16, 32));
    const float mnew  = fmaxf(mrun, tmax);
    const float alpha = __expf(mrun - mnew);
    mrun = mnew;
    float rsum = 0.0f;
#pragma unroll
    for (int jb = 0; jb < 2; ++jb)
#pragma unroll
      for (int g = 0; g < 8; ++g) {
        const float p = __expf(st[jb][g] - mnew);
        st[jb][g] = p;
        rsum += p;
      }
    rsum += __shfl_xor(rsum, 16, 32);
    lrun = lrun * alpha + rsum;
#pragma unroll
    for (int t = 0; t < 4; ++t)
#pragma unroll
      for (int g = 0; g < 8; ++g) o[t][g] *= alpha;

    // --- build P^T B-operand: half-wave exchange only ---
    v16h pop;
#pragma unroll
    for (int g = 0; g < 8; ++g) {
      const float x0 = __shfl_xor(st[0][g], 16, 32);
      const float x1 = __shfl_xor(st[1][g], 16, 32);
      pop[g]     = (_Float16)(lhi ? x1 : st[0][g]);
      pop[g + 8] = (_Float16)(lhi ? st[1][g] : x0);
    }
    // --- O^T += V^T * P^T (4 d-blocks of 16) ---
#pragma unroll
    for (int db = 0; db < 4; ++db) {
      const _Float16* p = vB + (size_t)(db * 16 + l16) * S_ + j0 + lhi * 16;
      v8h lo = *(const v8h*)p;
      v8h hi = *(const v8h*)(p + 8);
      v16h a;
#pragma unroll
      for (int t = 0; t < 8; ++t) { a[t] = lo[t]; a[t + 8] = hi[t]; }
      o[db] = __builtin_amdgcn_wmma_f32_16x16x32_f16(
          false, a, false, pop, (short)0, o[db], false, false);
    }
  }

  // Epilogue: normalize, f16, packed b128 stores (lane = s, 8 consecutive d).
  const float inv = 1.0f / lrun;
  _Float16* orow = xo + ((size_t)b * S_ + (s0 + l16)) * D_ + h * DK_;
#pragma unroll
  for (int db = 0; db < 4; ++db) {
    v8h pk;
#pragma unroll
    for (int g = 0; g < 8; ++g) pk[g] = (_Float16)(o[db][g] * inv);
    *(v8h*)(orow + db * 16 + lhi * 8) = pk;
  }
}

// ---------------------------------------------------------------------------
extern "C" void kernel_launch(void* const* d_in, const int* in_sizes, int n_in,
                              void* d_out, int out_size, void* d_ws,
                              size_t ws_size, hipStream_t stream) {
  const float* Q  = (const float*)d_in[0];
  const float* K  = (const float*)d_in[1];
  const float* V  = (const float*)d_in[2];
  const float* Wq = (const float*)d_in[3];
  const float* bq = (const float*)d_in[4];
  const float* Wk = (const float*)d_in[5];
  const float* bk = (const float*)d_in[6];
  const float* Wv = (const float*)d_in[7];
  const float* bv = (const float*)d_in[8];
  const float* Wo = (const float*)d_in[9];
  const float* bo = (const float*)d_in[10];

  const size_t BSD = (size_t)B_ * S_ * D_;  // 4,194,304
  const size_t DD  = (size_t)D_ * D_;       // 1,048,576

  _Float16* ws  = (_Float16*)d_ws;
  _Float16* Qh  = ws;               ws += BSD;
  _Float16* Kh  = ws;               ws += BSD;
  _Float16* Vh  = ws;               ws += BSD;
  _Float16* Wqh = ws;               ws += DD;
  _Float16* Wkh = ws;               ws += DD;
  _Float16* Wvh = ws;               ws += DD;
  _Float16* Woh = ws;               ws += DD;
  _Float16* qp  = ws;               ws += BSD;  // [B,H,S,DK]
  _Float16* kp  = ws;               ws += BSD;  // [B,H,S,DK]
  _Float16* vtp = ws;               ws += BSD;  // [B,H,DK,S]
  _Float16* xa  = ws;               ws += BSD;  // [B,S,D] attention out

  // 1) fp32 -> f16 casts
  {
    const int threads = 256;
    int n4 = (int)(BSD / 4);
    int blk = (n4 + threads - 1) / threads;
    mha_cvt_f16<<<blk, threads, 0, stream>>>(Q, Qh, n4);
    mha_cvt_f16<<<blk, threads, 0, stream>>>(K, Kh, n4);
    mha_cvt_f16<<<blk, threads, 0, stream>>>(V, Vh, n4);
    int w4 = (int)(DD / 4);
    int wblk = (w4 + threads - 1) / threads;
    mha_cvt_f16<<<wblk, threads, 0, stream>>>(Wq, Wqh, w4);
    mha_cvt_f16<<<wblk, threads, 0, stream>>>(Wk, Wkh, w4);
    mha_cvt_f16<<<wblk, threads, 0, stream>>>(Wv, Wvh, w4);
    mha_cvt_f16<<<wblk, threads, 0, stream>>>(Wo, Woh, w4);
  }

  // 2) projections: 64x32 tile per wave, 8 waves per block
  const int gemmWaves  = (M_ / 64) * (D_ / 32);  // 2048
  const int gemmBlocks = gemmWaves / 8;          // 256
  mha_gemm_tn<<<gemmBlocks, 256, 0, stream>>>(Qh, Wqh, bq, qp,  0, M_, D_, D_);
  mha_gemm_tn<<<gemmBlocks, 256, 0, stream>>>(Kh, Wkh, bk, kp,  0, M_, D_, D_);
  mha_gemm_tn<<<gemmBlocks, 256, 0, stream>>>(Vh, Wvh, bv, vtp, 1, M_, D_, D_);

  // 3) attention: one wave per 16 q-rows -> 4096 waves -> 512 blocks
  const int attnWaves  = B_ * H_ * (S_ / 16);
  const int attnBlocks = attnWaves / 8;
  mha_attn<<<attnBlocks, 256, 0, stream>>>(qp, kp, vtp, xa);

  // 4) output projection -> fp32 d_out
  mha_gemm_tn<<<gemmBlocks, 256, 0, stream>>>(xa, Woh, bo, d_out, 2, M_, D_, D_);
}
